// EncoderVidCRN_16776142258298
// MI455X (gfx1250) — matrix operand order repaired and checked by reading, not compiled, not run on hardware
//
#include <hip/hip_runtime.h>

// ---------------------------------------------------------------------------
// EncoderVidCRN for MI455X (gfx1250).
// All GEMMs: v_wmma_f32_16x16x32_bf16, operands pre-converted to bf16,
// LDS tiles staged with global_load_async_to_lds_b128 (ASYNCcnt-tracked,
// double buffered), f32 accumulate + fused epilogues.
// ---------------------------------------------------------------------------

typedef __attribute__((ext_vector_type(16))) __bf16 v16bf;
typedef __attribute__((ext_vector_type(8)))  float  v8f;

#define BMT 128
#define BNT 128
#define BKT 32
#define LDR 40            // padded LDS row stride (bf16 elems) -> conflict-free frags
#define TILE_E (BMT*LDR)  // 5120 bf16 per matrix tile
#define BUF_E (2*TILE_E)  // A tile + B tile per buffer

union ABfrag { v16bf v; uint4 q[2]; };

__device__ __forceinline__ unsigned pk_bf16(float x, float y) {
  union { __bf16 h[2]; unsigned u; } t;
  t.h[0] = (__bf16)x; t.h[1] = (__bf16)y;
  return t.u;
}

// CDNA5 async copy: global memory -> LDS, inst offset added to both addresses (16B/lane)
__device__ __forceinline__ void async_b128(unsigned lds, const __bf16* g) {
  asm volatile("global_load_async_to_lds_b128 %0, %1, off"
               :: "v"(lds), "v"(g) : "memory");
}
__device__ __forceinline__ void async_b128_o16(unsigned lds, const __bf16* g) {
  asm volatile("global_load_async_to_lds_b128 %0, %1, off offset:16"
               :: "v"(lds), "v"(g) : "memory");
}

// ---------------------------------------------------------------------------
// out = epilogue(A[M,K](bf16) @ W[N,K](bf16)^T + bias[N](f32))
//   MODE 0: out = acc + bias
//   MODE 1: out = elu(acc + bias)
//   MODE 2: out *= sigmoid(acc + bias)
// Batched over blockIdx.z. Out addressing:
//   idx = z*oZ + (r/oDiv)*oA + (r%oDiv)*oB + col  (standard: oDiv=1,oA=N,oB=0)
// Requires M%128==0, N%128==0, K%32==0 (true for every stage here).
// ---------------------------------------------------------------------------
template<int MODE>
__global__ __launch_bounds__(256)
void gemm_wmma(const __bf16* __restrict__ A, long sAz, int K,
               const __bf16* __restrict__ W, long sWz,
               const float* __restrict__ bias, long sBiasZ,
               float* __restrict__ out, long oZ, long oA, long oB, int oDiv) {
  __shared__ __bf16 smem[2 * BUF_E];  // double-buffered {A,B} tiles (40 KB)

  const int tid  = threadIdx.x;
  const int lane = tid & 31;
  const int wid  = tid >> 5;   // 8 wave32 waves
  const int wm   = wid >> 1;   // 0..3 : 32-row strip
  const int wn   = wid & 1;    // 0..1 : 64-col strip
  const int lm   = lane & 15;
  const int lh   = lane >> 4;
  const int z    = blockIdx.z;

  const __bf16* Ab = A + (long)z * sAz + (long)blockIdx.y * BMT * K;
  const __bf16* Wb = W + (long)z * sWz + (long)blockIdx.x * BNT * K;

  const int srow  = tid >> 1;  // staging: 128 rows, 2 threads/row, 32B each
  const int shalf = tid & 1;

  const unsigned ldsBase = (unsigned)(size_t)(void*)smem;
  const unsigned stA = (unsigned)((srow * LDR + shalf * 16) * 2);          // bytes
  const unsigned stB = (unsigned)((TILE_E + srow * LDR + shalf * 16) * 2); // bytes

  v8f acc[2][4] = {};

  // prologue: stage tile 0 into buffer 0 (4 async instructions / wave)
  {
    const __bf16* ga = Ab + (long)srow * K + shalf * 16;
    const __bf16* gw = Wb + (long)srow * K + shalf * 16;
    async_b128(ldsBase + stA, ga); async_b128_o16(ldsBase + stA, ga);
    async_b128(ldsBase + stB, gw); async_b128_o16(ldsBase + stB, gw);
  }

  int cur = 0;
  for (int k0 = 0; k0 < K; k0 += BKT, cur ^= 1) {
    if (k0 + BKT < K) {   // stage next tile into the other buffer, then wait
      const unsigned bo = (unsigned)((cur ^ 1) * BUF_E * 2);
      const __bf16* ga = Ab + (long)srow * K + (k0 + BKT) + shalf * 16;
      const __bf16* gw = Wb + (long)srow * K + (k0 + BKT) + shalf * 16;
      async_b128(ldsBase + bo + stA, ga); async_b128_o16(ldsBase + bo + stA, ga);
      async_b128(ldsBase + bo + stB, gw); async_b128_o16(ldsBase + bo + stB, gw);
      if (k0 + 2 * BKT < K) {
        __builtin_prefetch(ga + BKT, 0, 1);   // global_prefetch_b8, tile after next
        __builtin_prefetch(gw + BKT, 0, 1);
      }
      asm volatile("s_wait_asynccnt 0x4" ::: "memory");  // current tile landed
    } else {
      asm volatile("s_wait_asynccnt 0x0" ::: "memory");
    }
    __syncthreads();   // all waves' async data visible in LDS

    const __bf16* sA = smem + cur * BUF_E;
    const __bf16* sB = sA + TILE_E;

    // A frag (ISA 16-bit A layout): lane = M row lm, K = lh*8..+7 and 16+lh*8..+7
    ABfrag a[2];
#pragma unroll
    for (int i = 0; i < 2; ++i) {
      const __bf16* p = &sA[(wm * 32 + i * 16 + lm) * LDR + lh * 8];
      a[i].q[0] = *(const uint4*)(p);
      a[i].q[1] = *(const uint4*)(p + 16);
    }
    // B frag: lane = N row lm (row of W), K = lh*8..+7 and 16+lh*8..+7
    ABfrag b[4];
#pragma unroll
    for (int j = 0; j < 4; ++j) {
      const __bf16* p = &sB[(wn * 64 + j * 16 + lm) * LDR + lh * 8];
      b[j].q[0] = *(const uint4*)(p);
      b[j].q[1] = *(const uint4*)(p + 16);
    }
#pragma unroll
    for (int i = 0; i < 2; ++i)
#pragma unroll
      for (int j = 0; j < 4; ++j)
        acc[i][j] = __builtin_amdgcn_wmma_f32_16x16x32_bf16(
            false, a[i].v, false, b[j].v, (short)0, acc[i][j], false, false);

    __syncthreads();   // all waves done reading buf[cur] before it is re-staged
  }

  // epilogue: C/D layout: VGPR v, lane L -> row = 8*(L>>4)+v, col = L&15
  const int rowBase = blockIdx.y * BMT + wm * 32;
  const int colBase = blockIdx.x * BNT + wn * 64;
#pragma unroll
  for (int i = 0; i < 2; ++i) {
#pragma unroll
    for (int j = 0; j < 4; ++j) {
      const int col = colBase + j * 16 + lm;
      const float bv = bias[(long)z * sBiasZ + col];
#pragma unroll
      for (int v = 0; v < 8; ++v) {
        const int r = rowBase + i * 16 + lh * 8 + v;
        float val = acc[i][j][v] + bv;
        const int rq = r / oDiv;
        const int rm = r - rq * oDiv;
        const long idx = (long)z * oZ + (long)rq * oA + (long)rm * oB + col;
        if (MODE == 0) {
          out[idx] = val;
        } else if (MODE == 1) {
          out[idx] = val > 0.f ? val : (__expf(val) - 1.f);
        } else {
          out[idx] *= 1.f / (1.f + __expf(-val));
        }
      }
    }
  }
}

// ---------------------------------------------------------------------------
// f32 -> bf16 (4 elems / thread)
// ---------------------------------------------------------------------------
__global__ __launch_bounds__(256)
void to_bf16(const float* __restrict__ src, __bf16* __restrict__ dst, int n4) {
  const int i = blockIdx.x * 256 + threadIdx.x;
  if (i < n4) {
    float4 f = ((const float4*)src)[i];
    ((uint2*)dst)[i] = make_uint2(pk_bf16(f.x, f.y), pk_bf16(f.z, f.w));
  }
}

// ---------------------------------------------------------------------------
// CRN h-builder (bf16 out): h[z,r,0:512] = mean over chosen object subset,
// h[z,r,512:1024] = cond[r/condDiv].  objs element (r,o,d) at:
//   o*objStride + (r/rowDiv)*aStride + (r%rowDiv)*bStride + d
// Deterministic LCG subset stand-in for trace-time numpy sampling.
// ---------------------------------------------------------------------------
__device__ __forceinline__ unsigned choose_mask(unsigned seed, int n, int k) {
  unsigned mask = 0; int c = 0; unsigned s = seed | 1u;
  while (c < k) {
    s = s * 1664525u + 1013904223u;
    int i = (int)((s >> 8) % (unsigned)n);
    if (!((mask >> i) & 1u)) { mask |= 1u << i; ++c; }
  }
  return mask;
}

__global__ __launch_bounds__(256)
void build_h(const float* __restrict__ objs, const float* __restrict__ cond,
             __bf16* __restrict__ h, int n, long objStride, long aStride,
             long bStride, int rowDiv, int condDiv, long hStrideZ,
             unsigned seedBase) {
  const int z = blockIdx.z;
  const int r = blockIdx.y;
  const int d = blockIdx.x * 256 + threadIdx.x;  // 0..511
  const int scale = n - (z + 1);
  const unsigned mask = choose_mask(seedBase + (unsigned)z * 2654435761u, n, scale);
  const int rq = r / rowDiv;
  const int rm = r - rq * rowDiv;
  const long base = (long)rq * aStride + (long)rm * bStride + d;
  float s = 0.f;
  int cnt = 0;
#pragma unroll 4
  for (int o = 0; o < n; ++o)
    if ((mask >> o) & 1u) { s += objs[base + (long)o * objStride]; ++cnt; }
  __bf16* hz = h + (long)z * hStrideZ + (long)r * 1024;
  hz[d]       = (__bf16)(s / (float)cnt);
  hz[512 + d] = (__bf16)cond[(long)(r / condDiv) * 512 + d];
}

// ---------------------------------------------------------------------------
// LSTM cell (one step); gates_x precomputed for all t: [B*8, 2048] f32
// h state written as bf16 (feeds the W_hh / Wvm WMMA GEMMs directly)
// ---------------------------------------------------------------------------
__global__ __launch_bounds__(256)
void lstm_cell(const float* __restrict__ gx, const float* __restrict__ gh,
               float* __restrict__ c, __bf16* __restrict__ h, int t) {
  const int idx = blockIdx.x * 256 + threadIdx.x;  // 128*512
  const int b = idx >> 9, d = idx & 511;
  const long xo = ((long)(b * 8 + t)) * 2048 + d;
  const long ho = (long)b * 2048 + d;
  const float gi = gx[xo]        + gh[ho];
  const float gf = gx[xo + 512]  + gh[ho + 512];
  const float gg = gx[xo + 1024] + gh[ho + 1024];
  const float go = gx[xo + 1536] + gh[ho + 1536];
  const float i_ = 1.f / (1.f + __expf(-gi));
  const float f_ = 1.f / (1.f + __expf(-gf));
  const float g_ = tanhf(gg);
  const float o_ = 1.f / (1.f + __expf(-go));
  const float cv = f_ * c[idx] + i_ * g_;
  c[idx] = cv;
  h[idx] = (__bf16)(o_ * tanhf(cv));
}

__global__ __launch_bounds__(256)
void zero_f32(float* __restrict__ p, int n) {
  const int i = blockIdx.x * 256 + threadIdx.x;
  if (i < n) p[i] = 0.f;
}

// ---------------------------------------------------------------------------
// Host side
// ---------------------------------------------------------------------------
static const long KB_ = 1024;

// --- bf16 weight arena (byte offsets) ---
static const size_t OW_Q   = 0;          //   262144 e
static const size_t OW_M   = 524288;     //  1048576 e
static const size_t OW_A   = 2621440;    //  1048576 e
static const size_t OW_VM  = 4718592;    //   262144 e
static const size_t OW_IH  = 5242880;    //  4194304 e
static const size_t OW_HH  = 13631488;   //  1048576 e
static const size_t OW_W1  = 15728640;   //  7864320 e
static const size_t OW_W2  = 31457280;   //  6815744 e
static const size_t OW_GW2 = 45088768;   //  6815744 e
static const size_t OW_W3  = 58720256;   //  3670016 e
static const size_t OW_W4  = 66060288;   //  2621440 e
static const size_t OW_GW4 = 71303168;   //  2621440 e  (ends 76546048)

// --- activations ---
static const size_t OFF_APPBF   = 76546048;   // app bf16 (67.1 MB); reused as bf16 h-arena
static const size_t OFF_MOTBF   = 143654912;  // 4.2 MB
static const size_t OFF_QBF     = 147849216;  // 131072 B
static const size_t OFF_ARENA2  = 147980288;  // 33.5 MB f32: appProj -> crn -> clip -> crn_vm
static const size_t OFF_MOTPROJ = 181534720;  // 2 MB f32
static const size_t OFF_QPROJ   = 183631872;  // 256 KB f32
static const size_t OFF_GX      = 183894016;  // 8 MB f32
static const size_t OFF_GH      = 192282624;  // 1 MB f32
static const size_t OFF_HSBF    = 193331200;  // 128 KB bf16
static const size_t OFF_CS      = 193462272;  // 256 KB f32 (contiguous after HSBF)
static const size_t OFF_VM      = 193724416;  // 256 KB f32

static void cvt(const float* s, __bf16* d, long n, hipStream_t st) {
  long n4 = n / 4;
  to_bf16<<<dim3((unsigned)((n4 + 255) / 256)), dim3(256), 0, st>>>(s, d, (int)n4);
}

template<int MODE>
static void gemmL(const __bf16* A, long sAz, int K, const __bf16* W, long sWz,
                  const float* bias, long sBz, float* out, long oZ, long oA,
                  long oB, int oDiv, int M, int N, int Z, hipStream_t st) {
  dim3 g(N / 128, M / 128, Z), b(256);
  gemm_wmma<MODE><<<g, b, 0, st>>>(A, sAz, K, W, sWz, bias, sBz, out, oZ, oA, oB, oDiv);
}

extern "C" void kernel_launch(void* const* d_in, const int* in_sizes, int n_in,
                              void* d_out, int out_size, void* d_ws, size_t ws_size,
                              hipStream_t stream) {
  (void)in_sizes; (void)n_in; (void)out_size; (void)ws_size;
  const float* app  = (const float*)d_in[0];   // [128,8,16,2048]
  const float* mot  = (const float*)d_in[1];   // [128,8,2048]
  const float* q    = (const float*)d_in[2];   // [128,512]
  const float* Wq   = (const float*)d_in[3];
  const float* bq   = (const float*)d_in[4];
  const float* Wm   = (const float*)d_in[5];
  const float* bm   = (const float*)d_in[6];
  const float* Wa   = (const float*)d_in[7];
  const float* ba   = (const float*)d_in[8];
  const float* Wvm  = (const float*)d_in[9];
  const float* bvm  = (const float*)d_in[10];
  const float* W_ih = (const float*)d_in[11];
  const float* W_hh = (const float*)d_in[12];
  const float* b_ih = (const float*)d_in[13];
  const float* b_hh = (const float*)d_in[14];
  const float* W1   = (const float*)d_in[15];
  const float* b1   = (const float*)d_in[16];
  const float* W2   = (const float*)d_in[17];
  const float* b2   = (const float*)d_in[18];
  const float* gW2  = (const float*)d_in[19];
  const float* gb2  = (const float*)d_in[20];
  const float* W3   = (const float*)d_in[21];
  const float* b3   = (const float*)d_in[22];
  const float* W4   = (const float*)d_in[23];
  const float* b4   = (const float*)d_in[24];
  const float* gW4  = (const float*)d_in[25];
  const float* gb4  = (const float*)d_in[26];

  char* ws = (char*)d_ws;
  __bf16* wqB  = (__bf16*)(ws + OW_Q);
  __bf16* wmB  = (__bf16*)(ws + OW_M);
  __bf16* waB  = (__bf16*)(ws + OW_A);
  __bf16* wvmB = (__bf16*)(ws + OW_VM);
  __bf16* wihB = (__bf16*)(ws + OW_IH);
  __bf16* whhB = (__bf16*)(ws + OW_HH);
  __bf16* w1B  = (__bf16*)(ws + OW_W1);
  __bf16* w2B  = (__bf16*)(ws + OW_W2);
  __bf16* gw2B = (__bf16*)(ws + OW_GW2);
  __bf16* w3B  = (__bf16*)(ws + OW_W3);
  __bf16* w4B  = (__bf16*)(ws + OW_W4);
  __bf16* gw4B = (__bf16*)(ws + OW_GW4);

  __bf16* appB    = (__bf16*)(ws + OFF_APPBF);
  __bf16* hbuf    = (__bf16*)(ws + OFF_APPBF);   // reused after app_proj GEMM
  __bf16* motB    = (__bf16*)(ws + OFF_MOTBF);
  __bf16* qB      = (__bf16*)(ws + OFF_QBF);
  float*  arena2  = (float*)(ws + OFF_ARENA2);   // appProj / crn / clip / crn_vm
  float*  motProj = (float*)(ws + OFF_MOTPROJ);
  float*  qProj   = (float*)(ws + OFF_QPROJ);
  float*  gx      = (float*)(ws + OFF_GX);
  float*  gh      = (float*)(ws + OFF_GH);
  __bf16* hsB     = (__bf16*)(ws + OFF_HSBF);
  float*  cs      = (float*)(ws + OFF_CS);
  float*  vm      = (float*)(ws + OFF_VM);
  float*  out     = (float*)d_out;               // [128, 48, 512]

  const long WS = 512 * KB_;  // per-scale weight stride (512*1024 elems)

  // ---- convert all GEMM operands to bf16 ----
  cvt(app, appB, 33554432, stream);
  cvt(mot, motB, 2097152, stream);
  cvt(q, qB, 65536, stream);
  cvt(Wq, wqB, 262144, stream);     cvt(Wm, wmB, 1048576, stream);
  cvt(Wa, waB, 1048576, stream);    cvt(Wvm, wvmB, 262144, stream);
  cvt(W_ih, wihB, 4194304, stream); cvt(W_hh, whhB, 1048576, stream);
  cvt(W1, w1B, 7864320, stream);    cvt(W2, w2B, 6815744, stream);
  cvt(gW2, gw2B, 6815744, stream);  cvt(W3, w3B, 3670016, stream);
  cvt(W4, w4B, 2621440, stream);    cvt(gW4, gw4B, 2621440, stream);

  // ---- projections ----
  gemmL<0>(qB, 0, 512, wqB, 0, bq, 0, qProj, 0, 512, 0, 1, 128, 512, 1, stream);
  gemmL<0>(appB, 0, 2048, waB, 0, ba, 0, arena2, 0, 512, 0, 1, 16384, 512, 1, stream);
  gemmL<0>(motB, 0, 2048, wmB, 0, bm, 0, motProj, 0, 512, 0, 1, 1024, 512, 1, stream);

  // ---- CRN stage 1: 14 scales over F=16 frames, cond = mot_proj ----
  {
    dim3 g(2, 1024, 14), b(256);
    build_h<<<g, b, 0, stream>>>(arena2, motProj, hbuf, 16,
                                 512, 8192, 0, 1, 1, 1024 * KB_, 0x1111u);
  }
  gemmL<1>(hbuf, 1024 * KB_, 1024, w1B + WS, WS, b1 + 512, 512,
           arena2, 512 * KB_, 512, 0, 1, 1024, 512, 14, stream);  // crn_m

  // ---- CRN stage 2 (gated): 12 scales over the 14 outputs, cond = q_proj ----
  {
    dim3 g(2, 1024, 12), b(256);
    build_h<<<g, b, 0, stream>>>(arena2, qProj, hbuf, 14,
                                 512 * KB_, 512, 0, 1, 8, 1024 * KB_, 0x2222u);
  }
  gemmL<1>(hbuf, 1024 * KB_, 1024, w2B + WS, WS, b2 + 512, 512,
           arena2, 512 * KB_, 512, 0, 1, 1024, 512, 12, stream);  // clip
  gemmL<2>(hbuf, 1024 * KB_, 1024, gw2B + WS, WS, gb2 + 512, 512,
           arena2, 512 * KB_, 512, 0, 1, 1024, 512, 12, stream);

  // ---- LSTM over motion (8 steps) ----
  gemmL<0>(motB, 0, 2048, wihB, 0, b_ih, 0, gx, 0, 2048, 0, 1, 1024, 2048, 1, stream);
  zero_f32<<<dim3(384), dim3(256), 0, stream>>>((float*)hsB, 98304);  // hsB + cs
  for (int t = 0; t < 8; ++t) {
    gemmL<0>(hsB, 0, 512, whhB, 0, b_hh, 0, gh, 0, 2048, 0, 1, 128, 2048, 1, stream);
    lstm_cell<<<dim3(256), dim3(256), 0, stream>>>(gx, gh, cs, hsB, t);
  }
  gemmL<0>(hsB, 0, 512, wvmB, 0, bvm, 0, vm, 0, 512, 0, 1, 128, 512, 1, stream);

  // ---- CRN stage 3: 6 scales over C=8 clips, cond = vm_proj ----
  // clip element [b,c,t,d] = clip[t*524288 + b*4096 + c*512 + d]; rows r=b*12+t
  {
    dim3 g(2, 1536, 6), b(256);
    build_h<<<g, b, 0, stream>>>(arena2, vm, hbuf, 8,
                                 512, 4096, 512 * KB_, 12, 12, 1536 * KB_, 0x3333u);
  }
  gemmL<1>(hbuf, 1536 * KB_, 1024, w3B + WS, WS, b3 + 512, 512,
           arena2, 768 * KB_, 512, 0, 1, 1536, 512, 6, stream);   // crn_vm

  // ---- CRN stage 4 (gated): 4 scales over the 6 outputs, cond = q_proj ----
  {
    dim3 g(2, 1536, 4), b(256);
    build_h<<<g, b, 0, stream>>>(arena2, qProj, hbuf, 6,
                                 768 * KB_, 6144, 512, 12, 12, 1536 * KB_, 0x4444u);
  }
  // d_out [b, z*12+t, d]: idx = z*6144 + (r/12)*24576 + (r%12)*512 + col
  gemmL<1>(hbuf, 1536 * KB_, 1024, w4B + WS, WS, b4 + 512, 512,
           out, 6144, 24576, 512, 12, 1536, 512, 4, stream);
  gemmL<2>(hbuf, 1536 * KB_, 1024, gw4B + WS, WS, gb4 + 512, 512,
           out, 6144, 24576, 512, 12, 1536, 512, 4, stream);
}
